// GATTransformerLinkPredictor_71279277245063
// MI455X (gfx1250) — matrix operand order repaired and checked by reading, not compiled, not run on hardware
//
#include <hip/hip_runtime.h>

#define NN    4096
#define INC   256
#define HID   128
#define H1N   4
#define NE    131072
#define NEP   131072
#define THD   4
#define TLY   2
#define FFD   512
#define DM    128

typedef __attribute__((ext_vector_type(16))) __bf16        v16bf;
typedef __attribute__((ext_vector_type(16))) unsigned short v16us;
typedef __attribute__((ext_vector_type(8)))  float          v8f;
typedef __attribute__((ext_vector_type(4)))  unsigned int   v4u;
typedef __attribute__((ext_vector_type(8)))  int            v8i;
typedef __attribute__((ext_vector_type(4)))  int            v4i;

// ---------------- bf16 helpers (RNE) ----------------
__device__ __forceinline__ unsigned short f2b(float f) {
    union { float f; unsigned u; } v; v.f = f;
    unsigned r = v.u + 0x7FFFu + ((v.u >> 16) & 1u);
    return (unsigned short)(r >> 16);
}
__device__ __forceinline__ float b2f(unsigned short s) {
    union { unsigned u; float f; } v; v.u = ((unsigned)s) << 16; return v.f;
}

// ---------------- WMMA fragment loaders (wave32, ISA 7.12.2 layouts) ----------------
// A 16x32 bf16 fragment from row-major [rows x pitch] matrix at (row0, k0).
// lanes 0-15: M=lane, K in {0..7,16..23}; lanes 16-31: M=lane-16, K in {8..15,24..31}
__device__ __forceinline__ v16bf frag_a(const unsigned short* __restrict__ p,
                                        int pitch, int row0, int k0) {
    int lane = threadIdx.x & 31, half = lane >> 4, m = lane & 15;
    const unsigned short* base = p + (size_t)(row0 + m) * pitch + k0;
    v16us u;
#pragma unroll
    for (int j = 0; j < 4; ++j) {
        int ka = half * 8 + 2 * j;
        int kb = 16 + half * 8 + 2 * j;
        u[2 * j]     = base[ka];
        u[2 * j + 1] = base[ka + 1];
        u[8 + 2 * j]     = base[kb];
        u[8 + 2 * j + 1] = base[kb + 1];
    }
    return __builtin_bit_cast(v16bf, u);
}

// B 32x16 bf16 fragment from B^T stored row-major [N x pitch(K)] at (n0, k0).
// lanes 0-15: N=lane, K in {0..7,16..23}; lanes 16-31: N=lane-16, K in {8..15,24..31}
__device__ __forceinline__ v16bf frag_bt(const unsigned short* __restrict__ p,
                                         int pitch, int n0, int k0) {
    int lane = threadIdx.x & 31, half = lane >> 4, n = lane & 15;
    const unsigned short* base = p + (size_t)(n0 + n) * pitch + k0;
    v16us u;
#pragma unroll
    for (int e = 0; e < 8; ++e) { u[e] = base[half * 8 + e]; u[8 + e] = base[16 + half * 8 + e]; }
    return __builtin_bit_cast(v16bf, u);
}

__device__ __forceinline__ v8f wmma_bf16(v16bf a, v16bf b, v8f c) {
    return __builtin_amdgcn_wmma_f32_16x16x32_bf16(false, a, false, b, (short)0, c,
                                                   false, false);
}

// ---------------- TDM: 2-D tile (rows x 32 bf16) global -> LDS, packed ----------------
// D# per CDNA5 ISA ch.8: group0 {count=1, lds_addr, global_addr, type=2},
// group1 {data_size=2B, tensor_dim0=K, tensor_dim1=N, tile_dim0=32, tile_dim1=rows,
//         tensor_dim0_stride=K}; groups 2/3 zero (2-D).
// This toolchain (clang-23/therock-10.0) uses the 6-arg builtin form.
__device__ __forceinline__ void tdm_load_w_tile(const unsigned short* gsrc, unsigned lds_off,
                                                int rows, int K, int N) {
    unsigned long long ga = (unsigned long long)(const void*)gsrc;
    v4u g0;
    g0[0] = 1u;                                            // count=1, user mode
    g0[1] = lds_off;                                       // LDS byte address
    g0[2] = (unsigned)(ga & 0xFFFFFFFFu);                  // global_addr[31:0]
    g0[3] = (unsigned)((ga >> 32) & 0x1FFFFFFu) | (2u << 30); // addr[56:32] | type=2
    v8i g1;
    g1[0] = (int)(1u << 16);                               // data_size=1 -> 2 bytes
    g1[1] = (int)((unsigned)K << 16);                      // tensor_dim0[15:0]
    g1[2] = (int)((unsigned)N << 16);                      // tensor_dim1[15:0]
    g1[3] = (int)(32u << 16);                              // tile_dim0 = 32 elements
    g1[4] = rows;                                          // tile_dim1 (tile_dim2=0)
    g1[5] = K;                                             // tensor_dim0_stride[31:0]
    g1[6] = 0; g1[7] = 0;
    v4i g2 = {0, 0, 0, 0};
    v4i g3 = {0, 0, 0, 0};
    v8i g4 = {0, 0, 0, 0, 0, 0, 0, 0};
    __builtin_amdgcn_tensor_load_to_lds(g0, g1, g2, g3, g4, 0);
}

// ---------------- small utility kernels ----------------
__global__ void cvt_kernel(const float* __restrict__ x, unsigned short* __restrict__ y, int n) {
    int i = blockIdx.x * blockDim.x + threadIdx.x;
    if (i < n) y[i] = f2b(x[i]);
}

__global__ void fill_kernel(unsigned* __restrict__ p, unsigned v, int n) {
    int i = blockIdx.x * blockDim.x + threadIdx.x;
    if (i < n) p[i] = v;
}

// ---------------- bf16 WMMA GEMM: C[M,N] = A[M,K] * W[N,K]^T ----------------
// 64x64 block tile, 4 waves each computing a 32x32 (2x2 WMMA) sub-tile.
// Double-buffered LDS; A staged via registers, W tile DMA'd by the Tensor Data Mover.
__global__ __launch_bounds__(128)
void gemm_kernel(const unsigned short* __restrict__ A, const unsigned short* __restrict__ W,
                 const float* __restrict__ bias, float* __restrict__ Cf,
                 unsigned short* __restrict__ Cb, int M, int N, int K, int relu) {
    __shared__ __align__(16) unsigned short sA[2][2048];   // [64 rows][32 k]
    __shared__ __align__(16) unsigned short sW[2][2048];
    const int tid = threadIdx.x, wave = tid >> 5, lane = tid & 31;
    const int half = lane >> 4, l15 = lane & 15;
    const int bm = blockIdx.x * 64, bn = blockIdx.y * 64;
    const int mi = (wave >> 1) * 32, ni = (wave & 1) * 32;

    uint4 ra[2];
    auto ld_a = [&](int k0) {
#pragma unroll
        for (int c = 0; c < 2; ++c) {
            int ch = tid * 2 + c, row = ch >> 2, off = (ch & 3) * 8;
            ra[c] = *(const uint4*)&A[(size_t)(bm + row) * K + k0 + off];
        }
    };
    auto st_a = [&](int p) {
#pragma unroll
        for (int c = 0; c < 2; ++c) {
            int ch = tid * 2 + c, row = ch >> 2, off = (ch & 3) * 8;
            *(uint4*)&sA[p][row * 32 + off] = ra[c];
        }
    };

    // prologue: tile 0
    if (wave == 0)
        tdm_load_w_tile(&W[(size_t)bn * K], (unsigned)(size_t)&sW[0][0], 64, K, N);
    ld_a(0);
    st_a(0);
    __builtin_amdgcn_s_wait_tensorcnt(0);
    __syncthreads();

    v8f acc[2][2] = {};
    int p = 0;
    for (int k0 = 0; k0 < K; k0 += 32) {
        const bool more = (k0 + 32) < K;
        if (more) {
            if (wave == 0)
                tdm_load_w_tile(&W[(size_t)bn * K + k0 + 32],
                                (unsigned)(size_t)&sW[p ^ 1][0], 64, K, N);
            ld_a(k0 + 32);
        }
        v16bf a0 = frag_a(sA[p], 32, mi, 0);
        v16bf a1 = frag_a(sA[p], 32, mi + 16, 0);
        v16bf b0 = frag_bt(sW[p], 32, ni, 0);
        v16bf b1 = frag_bt(sW[p], 32, ni + 16, 0);
        acc[0][0] = wmma_bf16(a0, b0, acc[0][0]);
        acc[0][1] = wmma_bf16(a0, b1, acc[0][1]);
        acc[1][0] = wmma_bf16(a1, b0, acc[1][0]);
        acc[1][1] = wmma_bf16(a1, b1, acc[1][1]);
        if (more) st_a(p ^ 1);
        __builtin_amdgcn_s_wait_tensorcnt(0);
        __syncthreads();
        p ^= 1;
    }

#pragma unroll
    for (int i = 0; i < 2; ++i)
#pragma unroll
        for (int j = 0; j < 2; ++j) {
            int col = bn + ni + j * 16 + l15;
            float bv = bias ? bias[col] : 0.f;
#pragma unroll
            for (int r = 0; r < 8; ++r) {
                int row = bm + mi + i * 16 + r + 8 * half;
                float v = acc[i][j][r] + bv;
                if (relu) v = fmaxf(v, 0.f);
                size_t idx = (size_t)row * N + col;
                if (Cf) Cf[idx] = v;
                if (Cb) Cb[idx] = f2b(v);
            }
        }
}

// Decoder GEMM, fused gather: A row e = concat(h[psrc[e]], h[pdst[e]]) (K=256), relu.
// Same 64x64 double-buffered scheme; A gather rows are dynamic so both operands use
// the register-staged path.
__global__ __launch_bounds__(128)
void gemm_gather_kernel(const unsigned short* __restrict__ Hb,
                        const int* __restrict__ psrc, const int* __restrict__ pdst,
                        const unsigned short* __restrict__ W, const float* __restrict__ bias,
                        unsigned short* __restrict__ Cb, int M, int N, int K) {
    __shared__ __align__(16) unsigned short sA[2][2048];
    __shared__ __align__(16) unsigned short sW[2][2048];
    const int tid = threadIdx.x, wave = tid >> 5, lane = tid & 31;
    const int half = lane >> 4, l15 = lane & 15;
    const int bm = blockIdx.x * 64, bn = blockIdx.y * 64;
    const int mi = (wave >> 1) * 32, ni = (wave & 1) * 32;

    uint4 ra[2], rw[2];
    auto ld = [&](int k0) {
#pragma unroll
        for (int c = 0; c < 2; ++c) {
            int ch = tid * 2 + c, row = ch >> 2, off = (ch & 3) * 8;
            int col = k0 + off;
            int node = (col < DM) ? psrc[bm + row] : pdst[bm + row];
            ra[c] = *(const uint4*)&Hb[(size_t)node * DM + (col & (DM - 1))];
            rw[c] = *(const uint4*)&W[(size_t)(bn + row) * K + k0 + off];
        }
    };
    auto st = [&](int p) {
#pragma unroll
        for (int c = 0; c < 2; ++c) {
            int ch = tid * 2 + c, row = ch >> 2, off = (ch & 3) * 8;
            *(uint4*)&sA[p][row * 32 + off] = ra[c];
            *(uint4*)&sW[p][row * 32 + off] = rw[c];
        }
    };

    ld(0); st(0);
    __syncthreads();

    v8f acc[2][2] = {};
    int p = 0;
    for (int k0 = 0; k0 < K; k0 += 32) {
        const bool more = (k0 + 32) < K;
        if (more) ld(k0 + 32);
        v16bf a0 = frag_a(sA[p], 32, mi, 0);
        v16bf a1 = frag_a(sA[p], 32, mi + 16, 0);
        v16bf b0 = frag_bt(sW[p], 32, ni, 0);
        v16bf b1 = frag_bt(sW[p], 32, ni + 16, 0);
        acc[0][0] = wmma_bf16(a0, b0, acc[0][0]);
        acc[0][1] = wmma_bf16(a0, b1, acc[0][1]);
        acc[1][0] = wmma_bf16(a1, b0, acc[1][0]);
        acc[1][1] = wmma_bf16(a1, b1, acc[1][1]);
        if (more) st(p ^ 1);
        __syncthreads();
        p ^= 1;
    }

#pragma unroll
    for (int i = 0; i < 2; ++i)
#pragma unroll
        for (int j = 0; j < 2; ++j) {
            int col = bn + ni + j * 16 + l15;
            float bv = bias ? bias[col] : 0.f;
#pragma unroll
            for (int r = 0; r < 8; ++r) {
                int row = bm + mi + i * 16 + r + 8 * half;
                float v = fmaxf(acc[i][j][r] + bv, 0.f);
                Cb[(size_t)row * N + col] = f2b(v);
            }
        }
}

// ---------------- GAT pieces ----------------
__global__ void gat_elr_kernel(const float* __restrict__ h, const float* __restrict__ al,
                               const float* __restrict__ ar, float* __restrict__ el,
                               float* __restrict__ er, int nh, int H, int F) {
    int i = blockIdx.x * blockDim.x + threadIdx.x;
    if (i >= nh) return;
    int hd = i % H;
    const float* hv = h + (size_t)i * F;
    const float* a = al + hd * F;
    const float* b = ar + hd * F;
    float sl = 0.f, sr = 0.f;
    for (int f = 0; f < F; ++f) { float x = hv[f]; sl += x * a[f]; sr += x * b[f]; }
    el[i] = sl; er[i] = sr;
}

__device__ __forceinline__ unsigned fkey(float x) {
    unsigned b = __float_as_uint(x);
    return (b & 0x80000000u) ? ~b : (b | 0x80000000u);
}
__device__ __forceinline__ float finv(unsigned k) {
    return __uint_as_float((k & 0x80000000u) ? (k & 0x7FFFFFFFu) : ~k);
}
#define NEG_INF_KEY 0x007FFFFFu   // fkey(-inf)

__global__ void edge_score_max_kernel(const int* __restrict__ src, const int* __restrict__ dst,
                                      const float* __restrict__ el, const float* __restrict__ er,
                                      float* __restrict__ ev, unsigned* __restrict__ mk,
                                      int E, int H) {
    int i = blockIdx.x * blockDim.x + threadIdx.x;
    if (i >= E * H) return;
    int e = i / H, hh = i - e * H;
    int s = src[e], d = dst[e];
    float x = el[s * H + hh] + er[d * H + hh];
    x = x > 0.f ? x : 0.2f * x;          // leaky relu
    ev[i] = x;
    atomicMax(&mk[d * H + hh], fkey(x));
}

__global__ void edge_exp_sum_kernel(const int* __restrict__ dst, const float* __restrict__ ev,
                                    const unsigned* __restrict__ mk, float* __restrict__ ex,
                                    float* __restrict__ ss, int E, int H) {
    int i = blockIdx.x * blockDim.x + threadIdx.x;
    if (i >= E * H) return;
    int e = i / H, hh = i - e * H;
    int d = dst[e];
    float v = __expf(ev[i] - finv(mk[d * H + hh]));
    ex[i] = v;
    atomicAdd(&ss[d * H + hh], v);
}

__global__ void edge_aggregate_kernel(const int* __restrict__ src, const int* __restrict__ dst,
                                      const float* __restrict__ h, const float* __restrict__ ex,
                                      const float* __restrict__ ss, float* __restrict__ out,
                                      int E, int H, int F) {
    int i = blockIdx.x * blockDim.x + threadIdx.x;
    if (i >= E * F) return;
    int e = i / F, f = i - e * F;
    int s = src[e], d = dst[e];
    for (int hh = 0; hh < H; ++hh) {
        float a = ex[e * H + hh] / ss[d * H + hh];
        atomicAdd(&out[((size_t)d * H + hh) * F + f], a * h[((size_t)s * H + hh) * F + f]);
    }
}

__global__ void bias_act_kernel(const float* __restrict__ x, const float* __restrict__ bias,
                                float* __restrict__ yf, unsigned short* __restrict__ yb,
                                int total, int C, int relu) {
    int i = blockIdx.x * blockDim.x + threadIdx.x;
    if (i >= total) return;
    float v = x[i] + (bias ? bias[i % C] : 0.f);
    if (relu) v = fmaxf(v, 0.f);
    if (yf) yf[i] = v;
    if (yb) yb[i] = f2b(v);
}

// ---------------- flash attention: 4 heads, hd=32, seq=4096 ----------------
__global__ __launch_bounds__(128)
void attn_kernel(const unsigned short* __restrict__ qkv, unsigned short* __restrict__ ao, int n) {
    __shared__ __align__(16) unsigned short sK[1024];    // [key 32][hd 32]
    __shared__ __align__(16) unsigned short sVt[1024];   // [hd 32][key 32]
    __shared__ __align__(16) unsigned short sP[4][512];  // per-wave P tile [16][32]
    const int tid = threadIdx.x, wave = tid >> 5, lane = tid & 31;
    const int half = lane >> 4, l15 = lane & 15;
    const int h = blockIdx.y;
    const int q0 = blockIdx.x * 64 + wave * 16;
    const int P3 = 3 * DM;

    v16bf aq = frag_a(qkv + h * 32, P3, q0, 0);   // Q fragment, reused every tile

    float mr[8], lr[8];
    v8f oa = {}, ob = {};
#pragma unroll
    for (int r = 0; r < 8; ++r) { mr[r] = -3.0e38f; lr[r] = 0.f; }
    const float scale = 0.17677669529663687f;     // 1/sqrt(32)

    const int ldr = tid >> 2, ldo = (tid & 3) * 8;
    for (int j0 = 0; j0 < n; j0 += 32) {
        *(uint4*)&sK[ldr * 32 + ldo] =
            *(const uint4*)&qkv[(size_t)(j0 + ldr) * P3 + DM + h * 32 + ldo];
        {
            const unsigned short* vr = &qkv[(size_t)(j0 + ldr) * P3 + 2 * DM + h * 32 + ldo];
#pragma unroll
            for (int t = 0; t < 8; ++t) sVt[(ldo + t) * 32 + ldr] = vr[t];
        }
        __syncthreads();
        v8f s0 = {}, s1 = {};
        s0 = wmma_bf16(aq, frag_bt(sK, 32, 0, 0), s0);    // keys j0..j0+15
        s1 = wmma_bf16(aq, frag_bt(sK, 32, 16, 0), s1);   // keys j0+16..j0+31
#pragma unroll
        for (int r = 0; r < 8; ++r) {
            float x0 = s0[r] * scale, x1 = s1[r] * scale;
            float t = fmaxf(x0, x1);
#pragma unroll
            for (int o = 8; o; o >>= 1) t = fmaxf(t, __shfl_xor(t, o, 32));
            float nm = fmaxf(mr[r], t);
            float corr = __expf(mr[r] - nm);
            float p0 = __expf(x0 - nm), p1 = __expf(x1 - nm);
            float ps = p0 + p1;
#pragma unroll
            for (int o = 8; o; o >>= 1) ps += __shfl_xor(ps, o, 32);
            lr[r] = lr[r] * corr + ps;
            mr[r] = nm;
            oa[r] *= corr; ob[r] *= corr;
            int m = r + 8 * half;
            sP[wave][m * 32 + l15]      = f2b(p0);
            sP[wave][m * 32 + 16 + l15] = f2b(p1);
        }
        __syncthreads();
        v16bf ap = frag_a(sP[wave], 32, 0, 0);
        oa = wmma_bf16(ap, frag_bt(sVt, 32, 0, 0), oa);
        ob = wmma_bf16(ap, frag_bt(sVt, 32, 16, 0), ob);
        __syncthreads();
    }
#pragma unroll
    for (int r = 0; r < 8; ++r) {
        int row = q0 + r + 8 * half;
        float inv = 1.0f / lr[r];
        ao[(size_t)row * DM + h * 32 + l15]      = f2b(oa[r] * inv);
        ao[(size_t)row * DM + h * 32 + 16 + l15] = f2b(ob[r] * inv);
    }
}

// ---------------- residual + layernorm (wave per row, D=128) ----------------
__global__ __launch_bounds__(128)
void resid_ln_kernel(const float* __restrict__ x, const float* __restrict__ r,
                     const float* __restrict__ g, const float* __restrict__ b,
                     float* __restrict__ yf, unsigned short* __restrict__ yb, int rows) {
    int wave = threadIdx.x >> 5, lane = threadIdx.x & 31;
    int row = blockIdx.x * 4 + wave;
    if (row >= rows) return;
    const float* xr = x + (size_t)row * DM;
    const float* rr = r + (size_t)row * DM;
    float v[4]; float s = 0.f;
#pragma unroll
    for (int i = 0; i < 4; ++i) { int c = lane + 32 * i; v[i] = xr[c] + rr[c]; s += v[i]; }
#pragma unroll
    for (int o = 16; o; o >>= 1) s += __shfl_xor(s, o, 32);
    float mu = s * (1.0f / DM);
    float q = 0.f;
#pragma unroll
    for (int i = 0; i < 4; ++i) { v[i] -= mu; q += v[i] * v[i]; }
#pragma unroll
    for (int o = 16; o; o >>= 1) q += __shfl_xor(q, o, 32);
    float inv = rsqrtf(q * (1.0f / DM) + 1e-5f);
#pragma unroll
    for (int i = 0; i < 4; ++i) {
        int c = lane + 32 * i;
        float o2 = v[i] * inv * g[c] + b[c];
        yf[(size_t)row * DM + c] = o2;
        yb[(size_t)row * DM + c] = f2b(o2);
    }
}

// ---------------- decoder head: sigmoid(x . fc2_w + fc2_b), wave per edge ----------------
__global__ __launch_bounds__(128)
void decoder_out_kernel(const unsigned short* __restrict__ x, const float* __restrict__ w,
                        const float* __restrict__ bias, float* __restrict__ out, int n) {
    int wave = threadIdx.x >> 5, lane = threadIdx.x & 31;
    int e = blockIdx.x * 4 + wave;
    if (e >= n) return;
    float s = 0.f;
#pragma unroll
    for (int i = 0; i < 4; ++i) { int c = lane + 32 * i; s += b2f(x[(size_t)e * HID + c]) * w[c]; }
#pragma unroll
    for (int o = 16; o; o >>= 1) s += __shfl_xor(s, o, 32);
    if (lane == 0) out[e] = 1.0f / (1.0f + __expf(-(s + bias[0])));
}

// =================================================================================
extern "C" void kernel_launch(void* const* d_in, const int* in_sizes, int n_in,
                              void* d_out, int out_size, void* d_ws, size_t ws_size,
                              hipStream_t stream) {
    (void)in_sizes; (void)n_in; (void)out_size; (void)ws_size;
    const float* features = (const float*)d_in[0];
    const int*   src  = (const int*)d_in[1];
    const int*   dst  = (const int*)d_in[2];
    const int*   psrc = (const int*)d_in[3];
    const int*   pdst = (const int*)d_in[4];
    const float* W1   = (const float*)d_in[5];
    const float* al1  = (const float*)d_in[6];
    const float* ar1  = (const float*)d_in[7];
    const float* b1   = (const float*)d_in[8];
    const float* W2   = (const float*)d_in[9];
    const float* al2  = (const float*)d_in[10];
    const float* ar2  = (const float*)d_in[11];
    const float* b2   = (const float*)d_in[12];
    const float* tw_qkv = (const float*)d_in[13];
    const float* tb_qkv = (const float*)d_in[14];
    const float* tw_o   = (const float*)d_in[15];
    const float* tb_o   = (const float*)d_in[16];
    const float* ln1_g  = (const float*)d_in[17];
    const float* ln1_b  = (const float*)d_in[18];
    const float* tw_ff1 = (const float*)d_in[19];
    const float* tb_ff1 = (const float*)d_in[20];
    const float* tw_ff2 = (const float*)d_in[21];
    const float* tb_ff2 = (const float*)d_in[22];
    const float* ln2_g  = (const float*)d_in[23];
    const float* ln2_b  = (const float*)d_in[24];
    const float* fc1_w  = (const float*)d_in[25];
    const float* fc1_b  = (const float*)d_in[26];
    const float* fc2_w  = (const float*)d_in[27];
    const float* fc2_b  = (const float*)d_in[28];
    float* out = (float*)d_out;

    char* wsb = (char*)d_ws;
    size_t off = 0;
    auto alloc = [&](size_t bytes) -> void* {
        void* p = wsb + off;
        off += (bytes + 255) & ~(size_t)255;
        return p;
    };

    unsigned short* feat_b = (unsigned short*)alloc((size_t)NN * INC * 2);
    unsigned short* W1b    = (unsigned short*)alloc((size_t)H1N * HID * INC * 2);
    unsigned short* W2b    = (unsigned short*)alloc((size_t)HID * H1N * HID * 2);
    unsigned short* wqkvb  = (unsigned short*)alloc((size_t)TLY * 3 * DM * DM * 2);
    unsigned short* wob    = (unsigned short*)alloc((size_t)TLY * DM * DM * 2);
    unsigned short* wf1b   = (unsigned short*)alloc((size_t)TLY * FFD * DM * 2);
    unsigned short* wf2b   = (unsigned short*)alloc((size_t)TLY * DM * FFD * 2);
    unsigned short* fc1b   = (unsigned short*)alloc((size_t)HID * 2 * HID * 2);
    float* h1   = (float*)alloc((size_t)NN * H1N * HID * 4);
    float* el1  = (float*)alloc((size_t)NN * H1N * 4);
    float* er1  = (float*)alloc((size_t)NN * H1N * 4);
    float* ev1  = (float*)alloc((size_t)NE * H1N * 4);
    float* ex1  = (float*)alloc((size_t)NE * H1N * 4);
    unsigned* mk1 = (unsigned*)alloc((size_t)NN * H1N * 4);
    float* ss1  = (float*)alloc((size_t)NN * H1N * 4);
    float* agg1 = (float*)alloc((size_t)NN * H1N * HID * 4);
    unsigned short* h1rb = (unsigned short*)alloc((size_t)NN * H1N * HID * 2);
    float* h2   = (float*)alloc((size_t)NN * HID * 4);
    float* el2  = (float*)alloc((size_t)NN * 4);
    float* er2  = (float*)alloc((size_t)NN * 4);
    float* ev2  = (float*)alloc((size_t)NE * 4);
    float* ex2  = (float*)alloc((size_t)NE * 4);
    unsigned* mk2 = (unsigned*)alloc((size_t)NN * 4);
    float* ss2  = (float*)alloc((size_t)NN * 4);
    float* agg2 = (float*)alloc((size_t)NN * HID * 4);
    float* hcur = (float*)alloc((size_t)NN * DM * 4);
    unsigned short* hcurb = (unsigned short*)alloc((size_t)NN * DM * 2);
    unsigned short* qkvb  = (unsigned short*)alloc((size_t)NN * 3 * DM * 2);
    unsigned short* attnb = (unsigned short*)alloc((size_t)NN * DM * 2);
    float* oproj = (float*)alloc((size_t)NN * DM * 4);
    unsigned short* ffb = (unsigned short*)alloc((size_t)NN * FFD * 2);
    float* ff2o = (float*)alloc((size_t)NN * DM * 4);
    unsigned short* xdec = (unsigned short*)alloc((size_t)NEP * HID * 2);

    dim3 T256(256), T128(128);
    auto G = [](int n, int b) { return dim3((unsigned)((n + b - 1) / b)); };

    // --- weight / feature conversion to bf16 ---
    cvt_kernel<<<G(NN * INC, 256), T256, 0, stream>>>(features, feat_b, NN * INC);
    cvt_kernel<<<G(H1N * HID * INC, 256), T256, 0, stream>>>(W1, W1b, H1N * HID * INC);
    cvt_kernel<<<G(HID * H1N * HID, 256), T256, 0, stream>>>(W2, W2b, HID * H1N * HID);
    cvt_kernel<<<G(TLY * 3 * DM * DM, 256), T256, 0, stream>>>(tw_qkv, wqkvb, TLY * 3 * DM * DM);
    cvt_kernel<<<G(TLY * DM * DM, 256), T256, 0, stream>>>(tw_o, wob, TLY * DM * DM);
    cvt_kernel<<<G(TLY * FFD * DM, 256), T256, 0, stream>>>(tw_ff1, wf1b, TLY * FFD * DM);
    cvt_kernel<<<G(TLY * DM * FFD, 256), T256, 0, stream>>>(tw_ff2, wf2b, TLY * DM * FFD);
    cvt_kernel<<<G(HID * 2 * HID, 256), T256, 0, stream>>>(fc1_w, fc1b, HID * 2 * HID);

    // --- GAT layer 1 ---
    gemm_kernel<<<dim3(NN / 64, (H1N * HID) / 64), T128, 0, stream>>>(
        feat_b, W1b, nullptr, h1, nullptr, NN, H1N * HID, INC, 0);
    gat_elr_kernel<<<G(NN * H1N, 128), T128, 0, stream>>>(h1, al1, ar1, el1, er1,
                                                          NN * H1N, H1N, HID);
    fill_kernel<<<G(NN * H1N * HID, 256), T256, 0, stream>>>((unsigned*)agg1, 0u, NN * H1N * HID);
    fill_kernel<<<G(NN * H1N, 256), T256, 0, stream>>>((unsigned*)ss1, 0u, NN * H1N);
    fill_kernel<<<G(NN * H1N, 256), T256, 0, stream>>>(mk1, NEG_INF_KEY, NN * H1N);
    edge_score_max_kernel<<<G(NE * H1N, 256), T256, 0, stream>>>(src, dst, el1, er1, ev1, mk1, NE, H1N);
    edge_exp_sum_kernel<<<G(NE * H1N, 256), T256, 0, stream>>>(dst, ev1, mk1, ex1, ss1, NE, H1N);
    edge_aggregate_kernel<<<G(NE * HID, 256), T256, 0, stream>>>(src, dst, h1, ex1, ss1, agg1,
                                                                 NE, H1N, HID);
    bias_act_kernel<<<G(NN * H1N * HID, 256), T256, 0, stream>>>(agg1, b1, nullptr, h1rb,
                                                                 NN * H1N * HID, H1N * HID, 1);

    // --- GAT layer 2 ---
    gemm_kernel<<<dim3(NN / 64, HID / 64), T128, 0, stream>>>(
        h1rb, W2b, nullptr, h2, nullptr, NN, HID, H1N * HID, 0);
    gat_elr_kernel<<<G(NN, 128), T128, 0, stream>>>(h2, al2, ar2, el2, er2, NN, 1, HID);
    fill_kernel<<<G(NN * HID, 256), T256, 0, stream>>>((unsigned*)agg2, 0u, NN * HID);
    fill_kernel<<<G(NN, 256), T256, 0, stream>>>((unsigned*)ss2, 0u, NN);
    fill_kernel<<<G(NN, 256), T256, 0, stream>>>(mk2, NEG_INF_KEY, NN);
    edge_score_max_kernel<<<G(NE, 256), T256, 0, stream>>>(src, dst, el2, er2, ev2, mk2, NE, 1);
    edge_exp_sum_kernel<<<G(NE, 256), T256, 0, stream>>>(dst, ev2, mk2, ex2, ss2, NE, 1);
    edge_aggregate_kernel<<<G(NE * HID, 256), T256, 0, stream>>>(src, dst, h2, ex2, ss2, agg2,
                                                                 NE, 1, HID);
    bias_act_kernel<<<G(NN * HID, 256), T256, 0, stream>>>(agg2, b2, hcur, hcurb,
                                                           NN * HID, HID, 0);

    // --- transformer encoder layers ---
    for (int L = 0; L < TLY; ++L) {
        gemm_kernel<<<dim3(NN / 64, (3 * DM) / 64), T128, 0, stream>>>(
            hcurb, wqkvb + (size_t)L * 3 * DM * DM, tb_qkv + L * 3 * DM,
            nullptr, qkvb, NN, 3 * DM, DM, 0);
        attn_kernel<<<dim3(NN / 64, THD), T128, 0, stream>>>(qkvb, attnb, NN);
        gemm_kernel<<<dim3(NN / 64, DM / 64), T128, 0, stream>>>(
            attnb, wob + (size_t)L * DM * DM, tb_o + L * DM, oproj, nullptr, NN, DM, DM, 0);
        resid_ln_kernel<<<dim3(NN / 4), T128, 0, stream>>>(
            oproj, hcur, ln1_g + L * DM, ln1_b + L * DM, hcur, hcurb, NN);
        gemm_kernel<<<dim3(NN / 64, FFD / 64), T128, 0, stream>>>(
            hcurb, wf1b + (size_t)L * FFD * DM, tb_ff1 + L * FFD, nullptr, ffb, NN, FFD, DM, 1);
        gemm_kernel<<<dim3(NN / 64, DM / 64), T128, 0, stream>>>(
            ffb, wf2b + (size_t)L * DM * FFD, tb_ff2 + L * DM, ff2o, nullptr, NN, DM, FFD, 0);
        resid_ln_kernel<<<dim3(NN / 4), T128, 0, stream>>>(
            ff2o, hcur, ln2_g + L * DM, ln2_b + L * DM, hcur, hcurb, NN);
    }

    // --- decoder ---
    gemm_gather_kernel<<<dim3(NEP / 64, HID / 64), T128, 0, stream>>>(
        hcurb, psrc, pdst, fc1b, fc1_b, xdec, NEP, HID, 2 * HID);
    decoder_out_kernel<<<dim3(NEP / 4), T128, 0, stream>>>(xdec, fc2_w, fc2_b, out, NEP);
}
